// QuantLinear_4724464026022
// MI455X (gfx1250) — matrix operand order repaired and checked by reading, not compiled, not run on hardware
//
#include <hip/hip_runtime.h>

typedef __attribute__((ext_vector_type(16))) _Float16 v16h;
typedef __attribute__((ext_vector_type(8)))  _Float16 v8h;
typedef __attribute__((ext_vector_type(8)))  float    v8f;
typedef __attribute__((ext_vector_type(4)))  float    v4f;

#define N_TOK  8192
#define IN_F   4096
#define OUT_F  4096
#define LDSS   72    // mid-tier/fallback row stride (64 halfs + 8 pad)

// FP4 (E2M1) code -> fp32 value, branch-free.
// LUT: [-6,-4,-3,-2,-1.5,-1,-0.5,0, 0,0.5,1,1.5,2,3,4,6]; LUT[c] = -LUT[15-c].
__device__ __forceinline__ float fp4_dec(unsigned c) {
    unsigned m   = (c & 8u) ? (c & 7u) : (7u - c);
    unsigned tab = (m & 4u) ? 0x46444240u : 0x3E3C3800u;
    unsigned hb  = (tab >> (8u * (m & 3u))) & 0xFFu;
    unsigned short bits = (unsigned short)((hb << 8) | ((c & 8u) ? 0u : 0x8000u));
    union { unsigned short u; _Float16 h; } cv; cv.u = bits;
    return (float)cv.h;
}

// ---------------- Pass 1a: dequant W4 -> f16, layout W_t[N][K] ----------------
__global__ __launch_bounds__(256)
void dequant_w_kernel(const int* __restrict__ qw, const float* __restrict__ scales,
                      _Float16* __restrict__ wt) {
    __shared__ _Float16 T[128 * 136];
    const int tid = threadIdx.x;
    const int n  = tid & 127;
    const int kq = tid >> 7;
    const int N0 = blockIdx.x * 128;
    const int K0 = blockIdx.y * 128;

    const float sc = scales[(size_t)(K0 >> 7) * OUT_F + N0 + n];
#pragma unroll
    for (int w = 0; w < 8; ++w) {
        int k8l = kq * 8 + w;
        unsigned word = (unsigned)__builtin_nontemporal_load(
            qw + (size_t)((K0 >> 3) + k8l) * OUT_F + N0 + n);
        v8h hv;
#pragma unroll
        for (int j = 0; j < 8; ++j)
            hv[j] = (_Float16)(fp4_dec((word >> (4 * j)) & 0xFu) * sc);
        *(v8h*)&T[n * 136 + k8l * 8] = hv;
    }
    __syncthreads();
#pragma unroll
    for (int i = 0; i < 4; ++i) {
        int s = tid + 256 * i;
        int r = s >> 3, c8 = (s & 7) << 3;
        *(v8h*)&wt[(size_t)(N0 + r) * IN_F + K0 + c8] = *(v8h*)&T[r * 136 + c8];
    }
}

// ---------------- Pass 1b: convert x fp32 -> f16 ----------------
__global__ __launch_bounds__(256)
void cvt_x_kernel(const float* __restrict__ x, _Float16* __restrict__ xh) {
    size_t base = ((size_t)blockIdx.x * 256 + threadIdx.x) * 8;
    v4f f0 = __builtin_nontemporal_load((const v4f*)(x + base));
    v4f f1 = __builtin_nontemporal_load((const v4f*)(x + base) + 1);
    v8h hv;
    hv[0] = (_Float16)f0[0]; hv[1] = (_Float16)f0[1];
    hv[2] = (_Float16)f0[2]; hv[3] = (_Float16)f0[3];
    hv[4] = (_Float16)f1[0]; hv[5] = (_Float16)f1[1];
    hv[6] = (_Float16)f1[2]; hv[7] = (_Float16)f1[3];
    *(v8h*)(xh + base) = hv;
}

// ---------------- Pass 2 (fast): async-to-LDS double-buffered f16 GEMM --------
// Tile 256(M) x 128(N), KCHUNK 32. 8 waves as 4(M) x 2(N); wave tile 64x64.
// LDS layout (bytes): A0 [0,20480) B0 [20480,30720) A1 [30720,51200) B1 [51200,61440)
// Row stride 40 halfs (80 B).
__global__ __launch_bounds__(256)
void gemm_async_kernel(const _Float16* __restrict__ xh, const _Float16* __restrict__ wt,
                       const float* __restrict__ bias, float* __restrict__ out) {
    __shared__ _Float16 smem[30720];   // 60 KB

    const int tid   = threadIdx.x;
    const int lane  = tid & 31;
    const int wave  = tid >> 5;
    const int waveM = wave >> 1;       // 0..3 -> 64 rows
    const int waveN = wave & 1;        // 0..1 -> 64 cols
    const int lr    = lane & 15;
    const int lhi   = lane >> 4;

    const int M0 = blockIdx.y * 256;
    const int N0 = blockIdx.x * 128;

    const unsigned smemB = (unsigned)(size_t)(&smem[0]);  // LDS byte address

    // async copy slots: s = tid + 256*i; row = s>>2; c8 = (s&3)*8 halfs (16B)
    const int rA = tid >> 2;
    const int c8 = (tid & 3) << 3;

    const _Float16* aBase = xh + (size_t)M0 * IN_F;
    const _Float16* bBase = wt + (size_t)N0 * IN_F;

    unsigned voffA[4], voffB[2];       // global byte offsets (advance 64 B/chunk)
    unsigned ldsA[4],  ldsB[2];        // LDS byte addresses (buffer 0)
#pragma unroll
    for (int i = 0; i < 4; ++i) {
        voffA[i] = (unsigned)((rA + 64 * i) * (IN_F * 2) + c8 * 2);
        ldsA[i]  = smemB + (unsigned)((rA + 64 * i) * 80 + c8 * 2);
    }
#pragma unroll
    for (int i = 0; i < 2; ++i) {
        voffB[i] = (unsigned)((rA + 64 * i) * (IN_F * 2) + c8 * 2);
        ldsB[i]  = smemB + 20480u + (unsigned)((rA + 64 * i) * 80 + c8 * 2);
    }

    auto issue = [&](unsigned bufByte) {
#pragma unroll
        for (int i = 0; i < 4; ++i) {
            asm volatile("global_load_async_to_lds_b128 %0, %1, %2"
                         :: "v"(ldsA[i] + bufByte), "v"(voffA[i]), "s"(aBase)
                         : "memory");
        }
#pragma unroll
        for (int i = 0; i < 2; ++i) {
            asm volatile("global_load_async_to_lds_b128 %0, %1, %2"
                         :: "v"(ldsB[i] + bufByte), "v"(voffB[i]), "s"(bBase)
                         : "memory");
        }
#pragma unroll
        for (int i = 0; i < 4; ++i) voffA[i] += 64u;
#pragma unroll
        for (int i = 0; i < 2; ++i) voffB[i] += 64u;
    };

    v8f acc[4][4];
#pragma unroll
    for (int i = 0; i < 4; ++i)
#pragma unroll
        for (int j = 0; j < 4; ++j)
#pragma unroll
            for (int r = 0; r < 8; ++r) acc[i][j][r] = 0.0f;

    auto compute = [&](int buf) {
        const _Float16* As = smem + buf * 15360;
        const _Float16* Bs = As + 10240;
        v16h a[4], b[4];
#pragma unroll
        for (int i = 0; i < 4; ++i) {
            // A 16x32 f16: lane<16 K{0..7,16..23}, lane>=16 K{8..15,24..31}
            int row = waveM * 64 + i * 16 + lr;
            const _Float16* p = &As[row * 40 + lhi * 8];
            ((v8h*)&a[i])[0] = *(const v8h*)p;
            ((v8h*)&a[i])[1] = *(const v8h*)(p + 16);
        }
#pragma unroll
        for (int j = 0; j < 4; ++j) {
            // B 32x16 f16: lane<16 K{0..7,16..23}, lane>=16 K{8..15,24..31}, N=lane&15
            int nrow = waveN * 64 + j * 16 + lr;
            const _Float16* p = &Bs[nrow * 40 + lhi * 8];
            ((v8h*)&b[j])[0] = *(const v8h*)p;
            ((v8h*)&b[j])[1] = *(const v8h*)(p + 16);
        }
#pragma unroll
        for (int i = 0; i < 4; ++i)
#pragma unroll
            for (int j = 0; j < 4; ++j)
                acc[i][j] = __builtin_amdgcn_wmma_f32_16x16x32_f16(
                    false, a[i], false, b[j], (short)0, acc[i][j], false, false);
    };

    issue(0u);                          // prologue: chunk 0 into buffer 0

    int buf = 0;
#pragma unroll 1
    for (int kb = 0; kb < IN_F - 32; kb += 32) {
        issue(buf ? 0u : 30720u);       // prefetch next chunk into other buffer
        asm volatile("s_wait_asynccnt 6" ::: "memory");   // current 6 done
        __syncthreads();                // whole tile resident
        compute(buf);
        __syncthreads();                // buffer free for next prefetch
        buf ^= 1;
    }
    // peeled final chunk (no prefetch)
    asm volatile("s_wait_asynccnt 0" ::: "memory");
    __syncthreads();
    compute(buf);

    // epilogue: C f32 layout VGPR r -> M = r + 8*lhi, N = lr; NT stores keep
    // the f16 working set resident in L2.
#pragma unroll
    for (int i = 0; i < 4; ++i) {
        int mrow = M0 + waveM * 64 + i * 16 + lhi * 8;
#pragma unroll
        for (int j = 0; j < 4; ++j) {
            int n = N0 + waveN * 64 + j * 16 + lr;
            float bvl = bias[n];
            float* po = out + (size_t)mrow * OUT_F + n;
#pragma unroll
            for (int r = 0; r < 8; ++r)
                __builtin_nontemporal_store(acc[i][j][r] + bvl, po + (size_t)r * OUT_F);
        }
    }
}

// ---------------- Pass 2 (mid-tier): A converted in-kernel --------------------
__global__ __launch_bounds__(256)
void gemm_f32a_kernel(const float* __restrict__ x, const _Float16* __restrict__ wt,
                      const float* __restrict__ bias, float* __restrict__ out) {
    __shared__ _Float16 As[256 * LDSS];
    __shared__ _Float16 Bs[128 * LDSS];

    const int tid   = threadIdx.x;
    const int lane  = tid & 31;
    const int wave  = tid >> 5;
    const int waveM = wave >> 1;
    const int waveN = wave & 1;
    const int lr    = lane & 15;
    const int lhi   = lane >> 4;
    const int M0 = blockIdx.y * 256;
    const int N0 = blockIdx.x * 128;

    v8f acc[4][4];
#pragma unroll
    for (int i = 0; i < 4; ++i)
#pragma unroll
        for (int j = 0; j < 4; ++j)
#pragma unroll
            for (int r = 0; r < 8; ++r) acc[i][j][r] = 0.0f;

#pragma unroll 1
    for (int kb = 0; kb < IN_F; kb += 64) {
        v4f xf[8][2];
#pragma unroll
        for (int i = 0; i < 8; ++i) {
            int s = tid + 256 * i;
            int r = s >> 3, c8 = (s & 7) << 3;
            const float* p = x + (size_t)(M0 + r) * IN_F + kb + c8;
            xf[i][0] = ((const v4f*)p)[0];
            xf[i][1] = ((const v4f*)p)[1];
        }
        v8h bv[4];
#pragma unroll
        for (int i = 0; i < 4; ++i) {
            int s = tid + 256 * i;
            int n = s >> 3, c8 = (s & 7) << 3;
            bv[i] = *(const v8h*)&wt[(size_t)(N0 + n) * IN_F + kb + c8];
        }
        __syncthreads();
#pragma unroll
        for (int i = 0; i < 8; ++i) {
            int s = tid + 256 * i;
            int r = s >> 3, c8 = (s & 7) << 3;
            v8h hv;
            hv[0] = (_Float16)xf[i][0][0]; hv[1] = (_Float16)xf[i][0][1];
            hv[2] = (_Float16)xf[i][0][2]; hv[3] = (_Float16)xf[i][0][3];
            hv[4] = (_Float16)xf[i][1][0]; hv[5] = (_Float16)xf[i][1][1];
            hv[6] = (_Float16)xf[i][1][2]; hv[7] = (_Float16)xf[i][1][3];
            *(v8h*)&As[r * LDSS + c8] = hv;
        }
#pragma unroll
        for (int i = 0; i < 4; ++i) {
            int s = tid + 256 * i;
            int n = s >> 3, c8 = (s & 7) << 3;
            *(v8h*)&Bs[n * LDSS + c8] = bv[i];
        }
        __syncthreads();

#pragma unroll
        for (int ks = 0; ks < 64; ks += 32) {
            v16h a[4], b[4];
#pragma unroll
            for (int i = 0; i < 4; ++i) {
                int row = waveM * 64 + i * 16 + lr;
                const _Float16* p = &As[row * LDSS + ks + lhi * 8];
                ((v8h*)&a[i])[0] = *(const v8h*)p;
                ((v8h*)&a[i])[1] = *(const v8h*)(p + 16);
            }
#pragma unroll
            for (int j = 0; j < 4; ++j) {
                int nrow = waveN * 64 + j * 16 + lr;
                const _Float16* p = &Bs[nrow * LDSS + ks + lhi * 8];
                ((v8h*)&b[j])[0] = *(const v8h*)p;
                ((v8h*)&b[j])[1] = *(const v8h*)(p + 16);
            }
#pragma unroll
            for (int i = 0; i < 4; ++i)
#pragma unroll
                for (int j = 0; j < 4; ++j)
                    acc[i][j] = __builtin_amdgcn_wmma_f32_16x16x32_f16(
                        false, a[i], false, b[j], (short)0, acc[i][j], false, false);
        }
    }
#pragma unroll
    for (int i = 0; i < 4; ++i) {
        int mrow = M0 + waveM * 64 + i * 16 + lhi * 8;
#pragma unroll
        for (int j = 0; j < 4; ++j) {
            int n = N0 + waveN * 64 + j * 16 + lr;
            float bvl = bias[n];
            float* po = out + (size_t)mrow * OUT_F + n;
#pragma unroll
            for (int r = 0; r < 8; ++r)
                __builtin_nontemporal_store(acc[i][j][r] + bvl, po + (size_t)r * OUT_F);
        }
    }
}

// ---------------- Fallback: all-in-one (small workspace) ----------------------
__global__ __launch_bounds__(256)
void qlinear_fused_kernel(const float* __restrict__ x, const int* __restrict__ qw,
                          const float* __restrict__ scales, const float* __restrict__ bias,
                          float* __restrict__ out) {
    __shared__ _Float16 As[128 * LDSS];
    __shared__ _Float16 Bs[128 * LDSS];

    const int tid = threadIdx.x;
    const int lane = tid & 31, wave = tid >> 5;
    const int waveM = wave >> 2, waveN = wave & 3;
    const int lr = lane & 15, lhi = lane >> 4;
    const int M0 = blockIdx.y * 128, N0 = blockIdx.x * 128;
    const int bn = tid & 127, bq = tid >> 7;

    v8f acc[4][2];
#pragma unroll
    for (int i = 0; i < 4; ++i)
#pragma unroll
        for (int j = 0; j < 2; ++j)
#pragma unroll
            for (int r = 0; r < 8; ++r) acc[i][j][r] = 0.0f;

#pragma unroll 1
    for (int kb = 0; kb < IN_F; kb += 64) {
        v4f xf[4][2];
#pragma unroll
        for (int i = 0; i < 4; ++i) {
            int s = tid + 256 * i;
            int r = s >> 3, c8 = (s & 7) << 3;
            const float* p = x + (size_t)(M0 + r) * IN_F + kb + c8;
            xf[i][0] = ((const v4f*)p)[0];
            xf[i][1] = ((const v4f*)p)[1];
        }
        int wword[4];
#pragma unroll
        for (int w = 0; w < 4; ++w)
            wword[w] = qw[(size_t)((kb >> 3) + bq + 2 * w) * OUT_F + N0 + bn];
        float sc = scales[(size_t)(kb >> 7) * OUT_F + N0 + bn];

        __syncthreads();
#pragma unroll
        for (int i = 0; i < 4; ++i) {
            int s = tid + 256 * i;
            int r = s >> 3, c8 = (s & 7) << 3;
            v8h hv;
            hv[0] = (_Float16)xf[i][0][0]; hv[1] = (_Float16)xf[i][0][1];
            hv[2] = (_Float16)xf[i][0][2]; hv[3] = (_Float16)xf[i][0][3];
            hv[4] = (_Float16)xf[i][1][0]; hv[5] = (_Float16)xf[i][1][1];
            hv[6] = (_Float16)xf[i][1][2]; hv[7] = (_Float16)xf[i][1][3];
            *(v8h*)&As[r * LDSS + c8] = hv;
        }
#pragma unroll
        for (int w = 0; w < 4; ++w) {
            int k8l = bq + 2 * w;
            unsigned word = (unsigned)wword[w];
            v8h wv;
#pragma unroll
            for (int j = 0; j < 8; ++j)
                wv[j] = (_Float16)(fp4_dec((word >> (4 * j)) & 0xFu) * sc);
            *(v8h*)&Bs[bn * LDSS + k8l * 8] = wv;
        }
        __syncthreads();

#pragma unroll
        for (int ks = 0; ks < 64; ks += 32) {
            v16h a[4], b[2];
#pragma unroll
            for (int i = 0; i < 4; ++i) {
                int row = waveM * 64 + i * 16 + lr;
                const _Float16* p = &As[row * LDSS + ks + lhi * 8];
                ((v8h*)&a[i])[0] = *(const v8h*)p;
                ((v8h*)&a[i])[1] = *(const v8h*)(p + 16);
            }
#pragma unroll
            for (int j = 0; j < 2; ++j) {
                int nrow = waveN * 32 + j * 16 + lr;
                const _Float16* p = &Bs[nrow * LDSS + ks + lhi * 8];
                ((v8h*)&b[j])[0] = *(const v8h*)p;
                ((v8h*)&b[j])[1] = *(const v8h*)(p + 16);
            }
#pragma unroll
            for (int i = 0; i < 4; ++i)
#pragma unroll
                for (int j = 0; j < 2; ++j)
                    acc[i][j] = __builtin_amdgcn_wmma_f32_16x16x32_f16(
                        false, a[i], false, b[j], (short)0, acc[i][j], false, false);
        }
    }
#pragma unroll
    for (int i = 0; i < 4; ++i) {
        int mrow = M0 + waveM * 64 + i * 16 + lhi * 8;
#pragma unroll
        for (int j = 0; j < 2; ++j) {
            int n = N0 + waveN * 32 + j * 16 + lr;
            float bvl = bias[n];
            float* po = out + (size_t)mrow * OUT_F + n;
#pragma unroll
            for (int r = 0; r < 8; ++r)
                __builtin_nontemporal_store(acc[i][j][r] + bvl, po + (size_t)r * OUT_F);
        }
    }
}

extern "C" void kernel_launch(void* const* d_in, const int* in_sizes, int n_in,
                              void* d_out, int out_size, void* d_ws, size_t ws_size,
                              hipStream_t stream) {
    const float* x      = (const float*)d_in[0];  // (8192, 4096) fp32
    const int*   qw     = (const int*)  d_in[1];  // (512, 4096) int32
    const float* scales = (const float*)d_in[2];  // (32, 4096) fp32
    const float* bias   = (const float*)d_in[3];  // (4096,) fp32
    float*       out    = (float*)d_out;

    const size_t needW = (size_t)OUT_F * IN_F * sizeof(_Float16);   // 32 MB
    const size_t needX = (size_t)N_TOK * IN_F * sizeof(_Float16);   // 64 MB

    if (ws_size >= needW + needX) {
        _Float16* wt = (_Float16*)d_ws;
        _Float16* xh = (_Float16*)((char*)d_ws + needW);
        dequant_w_kernel<<<dim3(OUT_F / 128, IN_F / 128), 256, 0, stream>>>(qw, scales, wt);
        cvt_x_kernel<<<(N_TOK * (size_t)IN_F) / (256 * 8), 256, 0, stream>>>(x, xh);
        gemm_async_kernel<<<dim3(OUT_F / 128, N_TOK / 256), 256, 0, stream>>>(
            xh, wt, bias, out);
    } else if (ws_size >= needW) {
        _Float16* wt = (_Float16*)d_ws;
        dequant_w_kernel<<<dim3(OUT_F / 128, IN_F / 128), 256, 0, stream>>>(qw, scales, wt);
        gemm_f32a_kernel<<<dim3(OUT_F / 128, N_TOK / 256), 256, 0, stream>>>(
            x, wt, bias, out);
    } else {
        qlinear_fused_kernel<<<dim3(OUT_F / 128, N_TOK / 128), 256, 0, stream>>>(
            x, qw, scales, bias, out);
    }
}
